// LSTM_p_ag_27212912787466
// MI455X (gfx1250) — matrix operand order repaired and checked by reading, not compiled, not run on hardware
//
#include <hip/hip_runtime.h>
#include <math.h>

// ---- problem constants (from reference) ----
constexpr int kH   = 128;          // lstm hidden
constexpr int kL   = 3;            // lstm layers
constexpr int kF   = 4096;         // factors
constexpr int kT   = 32;           // fn types
constexpr int kSL  = 32;           // nodes per formula (STATE_L)
constexpr int kB   = 16;           // batch of formulas
constexpr int kD   = 4160;         // F + FPL
constexpr int k2D  = 8320;         // 2*D
constexpr int kPVD = kF + kSL - 1; // 4127
constexpr int kP3  = 13;
constexpr float kParaScale = 4160.0f; // FPL + F
constexpr int kMaxN = 48;          // padded max state rows (actual max 33)
constexpr int kSeq  = kSL + 1;     // 33
constexpr int k4H  = 4 * kH;       // 512

// output tuple offsets in d_out (floats)
constexpr size_t OFF_FT = 0;
constexpr size_t OFF_I1 = (size_t)kB * kT;                 // 512
constexpr size_t OFF_I2 = OFF_I1 + (size_t)kB * kD;        // 67072
constexpr size_t OFF_M  = OFF_I2 + (size_t)kB * kD;        // 133632
constexpr size_t OFF_S  = OFF_M + (size_t)kB * kP3;        // 133840

// workspace offsets (floats)
constexpr size_t WS_SEQ = 0;                               // [B][33][H]
constexpr size_t WS_OUT = WS_SEQ + (size_t)kB * kSeq * kH; // [B][2H]
constexpr size_t WS_HID = WS_OUT + (size_t)kB * 2 * kH;    // [2][B][2D]
constexpr size_t WS_LOG = WS_HID + (size_t)2 * kB * k2D;   // [2][B][D]
constexpr size_t WS_PW  = WS_LOG + (size_t)2 * kB * kD;    // packed bf16 cell weights
// packed region: [L][2][kH][4H] bf16 = 393216 bf16 = 196608 floats

typedef __bf16 bf16;
typedef __attribute__((ext_vector_type(16))) __bf16 v16bf;
typedef __attribute__((ext_vector_type(8)))  float  v8f;
typedef __attribute__((ext_vector_type(2)))  float  f32x2;
typedef __attribute__((ext_vector_type(4)))  float  f32x4;
typedef __attribute__((ext_vector_type(4)))  unsigned int u32x4;
typedef __attribute__((ext_vector_type(8)))  int i32x8;
typedef __attribute__((ext_vector_type(4)))  int i32x4;

static __device__ __forceinline__ float sigmoidf_(float x) { return 1.0f / (1.0f + __expf(-x)); }

// CDNA5 ISA 7.12.2: 16-bit A-matrix 16x32 — element e of the v16bf for a lane
// in half `half` (lane>>4) maps to K = 16*(e>>3) + 8*half + 2*((e>>1)&3) + (e&1)
static __device__ __forceinline__ int a_kmap(int e, int half) {
  return ((e >> 3) << 4) + (half << 3) + (((e >> 1) & 3) << 1) + (e & 1);
}

// A fragment (16x32) from row-major f32 [rows x lda]; element pairs (2p,2p+1)
// are consecutive K -> 8x 8-byte loads. Rows >= mlim read as 0.
static __device__ __forceinline__ v16bf load_a_f32(const float* A, int lda, int m0, int k0,
                                                   int lane, int mlim) {
  const int half = lane >> 4;
  const int row  = m0 + (lane & 15);
  v16bf a;
  if (row < mlim) {
    const float* r = A + (size_t)row * lda + k0;
#pragma unroll
    for (int p = 0; p < 8; ++p) {
      const f32x2 v = *(const f32x2*)(r + a_kmap(2 * p, half));
      a[2 * p] = (bf16)v.x; a[2 * p + 1] = (bf16)v.y;
    }
  } else {
#pragma unroll
    for (int e = 0; e < 16; ++e) a[e] = (bf16)0.0f;
  }
  return a;
}

// A fragment where every row is the same vector (broadcast of `cur`)
static __device__ __forceinline__ v16bf load_a_bcast(const float* vec, int k0, int lane) {
  const int half = lane >> 4;
  v16bf a;
#pragma unroll
  for (int p = 0; p < 8; ++p) {
    const f32x2 v = *(const f32x2*)(vec + k0 + a_kmap(2 * p, half));
    a[2 * p] = (bf16)v.x; a[2 * p + 1] = (bf16)v.y;
  }
  return a;
}

// B fragment (32x16): lane%16 = N column, element e = K a_kmap(e, half); row-major f32 [K x ldb]
static __device__ __forceinline__ v16bf load_b_f32(const float* Bm, int ldb, int k0, int n0,
                                                   int lane) {
  const int half = lane >> 4;
  const float* c = Bm + (size_t)k0 * ldb + n0 + (lane & 15);
  v16bf b;
#pragma unroll
  for (int e = 0; e < 16; ++e) b[e] = (bf16)c[(size_t)a_kmap(e, half) * ldb];
  return b;
}

// Non-temporal variant for the once-streamed W2 (2x138MB > L2): keep the
// stream from evicting reused operands.
static __device__ __forceinline__ v16bf load_b_f32_nt(const float* Bm, int ldb, int k0, int n0,
                                                      int lane) {
  const int half = lane >> 4;
  const float* c = Bm + (size_t)k0 * ldb + n0 + (lane & 15);
  v16bf b;
#pragma unroll
  for (int e = 0; e < 16; ++e)
    b[e] = (bf16)__builtin_nontemporal_load(c + (size_t)a_kmap(e, half) * ldb);
  return b;
}

// B fragment from pre-packed bf16 [K=kH][N=4H] matrix: column n0+lane%16, element e = K a_kmap(e, half)
static __device__ __forceinline__ v16bf load_b_pk(const bf16* P, int k0, int n0, int lane) {
  const int half = lane >> 4;
  const bf16* c = P + (size_t)k0 * k4H + n0 + (lane & 15);
  v16bf b;
#pragma unroll
  for (int e = 0; e < 16; ++e) b[e] = c[(size_t)a_kmap(e, half) * k4H];
  return b;
}

static __device__ __forceinline__ v8f wmma_bf16(v16bf a, v16bf b, v8f c) {
  return __builtin_amdgcn_wmma_f32_16x16x32_bf16(false, a, false, b, (short)0, c, false, false);
}

// ============================================================================
// Kernel 0: transpose + convert the LSTM cell weights into WMMA-B bf16 layout
// packed[l][m][k][n] = W_m[l][n][k]  (m: 0=Wih, 1=Whh; [4H][H] -> [H][4H])
// ============================================================================
__global__ void pack_cell_weights_kernel(const float* __restrict__ Wih,
                                         const float* __restrict__ Whh,
                                         bf16* __restrict__ packed) {
  const size_t idx = (size_t)blockIdx.x * blockDim.x + threadIdx.x;
  if (idx >= (size_t)kL * 2 * kH * k4H) return;
  const int n = (int)(idx & (k4H - 1));
  const int k = (int)((idx >> 9) & (kH - 1));
  const int m = (int)((idx >> 16) & 1);
  const int l = (int)(idx >> 17);
  const float* W = m ? Whh : Wih;
  packed[idx] = (bf16)W[(size_t)l * k4H * kH + (size_t)n * kH + k];
}

// ============================================================================
// Kernel 1: chain-AST LSTM evaluation. One workgroup (8 wave32) per formula.
// Whole running (h, c) state for all 3 layers lives in LDS (MI455X: 320KB/WGP).
// Each wave owns a 16-row x 16-hidden-col tile and accumulates all 4 gates of
// g = X@Wih^T + H0@Whh^T in v8f accumulators via v_wmma_f32_16x16x32_bf16.
// ============================================================================
__global__ void chain_lstm_kernel(const float* __restrict__ emb_s, const float* __restrict__ emb_l,
                                  const float* __restrict__ fn_emb,
                                  const float* __restrict__ fus_W1, const float* __restrict__ fus_b1,
                                  const float* __restrict__ fus_W2, const float* __restrict__ fus_b2,
                                  const bf16* __restrict__ pw,    // packed [L][2][H][4H] bf16
                                  const float* __restrict__ cell_b,
                                  const long long* __restrict__ formulas, // int64 [B][32][7]
                                  float* __restrict__ seq_out)            // [B][33][H]
{
  extern __shared__ float smem[];
  float* s_st  = smem;                        // [L][MaxN][H] hidden state
  float* c_st  = s_st + kL * kMaxN * kH;      // [L][MaxN][H] cell state
  float* h_new = c_st + kL * kMaxN * kH;      // [MaxN][H]
  float* c_new = h_new + kMaxN * kH;          // [MaxN][H]
  float* cur   = c_new + kMaxN * kH;          // [H]
  float* t1    = cur + kH;                    // [2H] fusion hidden
  float* accs  = t1 + 2 * kH;                 // [SL] accumulated t_shift per node
  float* accp  = accs + kSL;                  // [SL] accumulated t_span per node

  const int bi = blockIdx.x;
  const int tid = threadIdx.x;
  const int wave = tid >> 5, lane = tid & 31;
  const int nwaves = (int)(blockDim.x >> 5);
  const long long* fm = formulas + (size_t)bi * kSL * 7;

  // accumulated (shift, span) down the chain from the root (node 31, (0,1))
  if (tid == 0) {
    float sh = 0.f, sp = 1.f;
    for (int i = kSL - 1; i >= 0; --i) {
      accs[i] = sh; accp[i] = sp;
      sh += (float)fm[i * 7 + 2];  // ts1 of node i applies to its child
      sp += (float)fm[i * 7 + 3];  // tp1
    }
  }
  __syncthreads();

  for (int node = 0; node < kSL; ++node) {
    const long long* f = fm + node * 7;
    const int t = (int)f[0];

    // append leaf embedding rows: node 0 contributes both leaves, others id2 only
    if (node == 0) {
      const int fac1 = -(int)f[1] - 1;
      for (int j = tid; j < kL * kH; j += blockDim.x) {
        const int l = j >> 7, hh = j & (kH - 1);
        s_st[(l * kMaxN + 0) * kH + hh] = emb_s[(size_t)fac1 * (kL * kH) + j];
        c_st[(l * kMaxN + 0) * kH + hh] = emb_l[(size_t)fac1 * (kL * kH) + j];
      }
    }
    {
      const int fac2 = -(int)f[4] - 1;
      const int row = node + 1;
      for (int j = tid; j < kL * kH; j += blockDim.x) {
        const int l = j >> 7, hh = j & (kH - 1);
        s_st[(l * kMaxN + row) * kH + hh] = emb_s[(size_t)fac2 * (kL * kH) + j];
        c_st[(l * kMaxN + row) * kH + hh] = emb_l[(size_t)fac2 * (kL * kH) + j];
      }
    }
    __syncthreads();

    // fusion MLP: cur = tanh([fn_emb[t], shift, span] @ fus_W1 + b1) @ fus_W2 + b2
    if (tid < 2 * kH) {
      float acc = fus_b1[tid];
      for (int k = 0; k < kH; ++k) acc += fn_emb[(size_t)t * kH + k] * fus_W1[k * (2 * kH) + tid];
      acc += accs[node] * fus_W1[kH * (2 * kH) + tid];
      acc += accp[node] * fus_W1[(kH + 1) * (2 * kH) + tid];
      t1[tid] = tanhf(acc);
    }
    __syncthreads();
    if (tid < kH) {
      float acc = fus_b2[tid];
      for (int k = 0; k < 2 * kH; ++k) acc += t1[k] * fus_W2[k * kH + tid];
      cur[tid] = acc;
    }
    __syncthreads();

    const int N = node + 2;
    const int mtiles = (N + 15) >> 4;

    for (int l = 0; l < kL; ++l) {
      const bf16* Pih = pw + (size_t)(l * 2 + 0) * kH * k4H;
      const bf16* Phh = pw + (size_t)(l * 2 + 1) * kH * k4H;
      const float* bl  = cell_b + (size_t)l * k4H;
      const float* Xin = (l == 0) ? cur : (s_st + (size_t)(l - 1) * kMaxN * kH);
      const float* H0  = s_st + (size_t)l * kMaxN * kH;
      const float* C0  = c_st + (size_t)l * kMaxN * kH;

      const int ntiles = mtiles * (kH / 16);
      for (int tile = wave; tile < ntiles; tile += nwaves) {
        const int m0  = (tile >> 3) << 4;  // state-row tile
        const int hn0 = (tile & 7) << 4;   // hidden-column tile (within H)
        v8f acc_i = {}, acc_f = {}, acc_g = {}, acc_o = {};
#pragma unroll
        for (int kk = 0; kk < kH / 32; ++kk) {
          const int k0 = kk * 32;
          v16bf ax = (l == 0) ? load_a_bcast(cur, k0, lane)
                              : load_a_f32(Xin, kH, m0, k0, lane, N);
          v16bf ah = load_a_f32(H0, kH, m0, k0, lane, N);
          v16bf b;
          b = load_b_pk(Pih, k0, 0 * kH + hn0, lane); acc_i = wmma_bf16(ax, b, acc_i);
          b = load_b_pk(Phh, k0, 0 * kH + hn0, lane); acc_i = wmma_bf16(ah, b, acc_i);
          b = load_b_pk(Pih, k0, 1 * kH + hn0, lane); acc_f = wmma_bf16(ax, b, acc_f);
          b = load_b_pk(Phh, k0, 1 * kH + hn0, lane); acc_f = wmma_bf16(ah, b, acc_f);
          b = load_b_pk(Pih, k0, 2 * kH + hn0, lane); acc_g = wmma_bf16(ax, b, acc_g);
          b = load_b_pk(Phh, k0, 2 * kH + hn0, lane); acc_g = wmma_bf16(ah, b, acc_g);
          b = load_b_pk(Pih, k0, 3 * kH + hn0, lane); acc_o = wmma_bf16(ax, b, acc_o);
          b = load_b_pk(Phh, k0, 3 * kH + hn0, lane); acc_o = wmma_bf16(ah, b, acc_o);
        }
        // gate nonlinearity directly on accumulator fragment (C/D layout 7.12.2)
        const int col = hn0 + (lane & 15);
        const int rbase = m0 + ((lane >> 4) << 3);
        const float bi_ = bl[0 * kH + col], bf_ = bl[1 * kH + col];
        const float bg_ = bl[2 * kH + col], bo_ = bl[3 * kH + col];
#pragma unroll
        for (int i = 0; i < 8; ++i) {
          const int r = rbase + i;
          if (r < N) {
            const float gi = acc_i[i] + bi_, gf = acc_f[i] + bf_;
            const float gg = acc_g[i] + bg_, go = acc_o[i] + bo_;
            const float cn = sigmoidf_(gf) * C0[(size_t)r * kH + col] + sigmoidf_(gi) * tanhf(gg);
            c_new[r * kH + col] = cn;
            h_new[r * kH + col] = sigmoidf_(go) * tanhf(cn);
          }
        }
      }
      __syncthreads();
      for (int j = tid; j < N * kH; j += blockDim.x) {
        s_st[(size_t)l * kMaxN * kH + j] = h_new[j];
        c_st[(size_t)l * kMaxN * kH + j] = c_new[j];
      }
      __syncthreads();
    }
  }

  // final-layer hidden rows [33][H] = sm[-1], the bi-LSTM input sequence
  for (int j = tid; j < kSeq * kH; j += blockDim.x)
    seq_out[(size_t)bi * kSeq * kH + j] = s_st[(size_t)(kL - 1) * kMaxN * kH + j];
}

// ============================================================================
// Kernel 2: 3-layer bi-LSTM over the 33-step sequence (batch 1 per block).
// Sequence staging global->LDS done by the Tensor Data Mover: a 33x128 f32
// tile with 128-dword LDS padding per row lands directly in the [33][2H]
// layout (D# per ISA 08 §8; pad_interval=128dw code 6, pad_amount=128dw).
// ============================================================================
__global__ void bilstm_kernel(const float* __restrict__ seq_in, // [B][33][H]
                              const float* __restrict__ Wih0, const float* __restrict__ Whh0,
                              const float* __restrict__ b0,
                              const float* __restrict__ Wih, const float* __restrict__ Whh,
                              const float* __restrict__ bb,
                              float* __restrict__ outv)          // [B][2H]
{
  extern __shared__ float smem[];
  float* xs   = smem;                   // [33][2H] layer input (layer0 uses cols 0..H-1)
  float* hf   = xs + kSeq * 2 * kH;     // [33][H]
  float* hb   = hf + kSeq * kH;         // [33][H]
  float* gb   = hb + kSeq * kH;         // [4H]
  float* hcur = gb + 4 * kH;            // [H]
  float* ccur = hcur + kH;              // [H]

  const int bi = blockIdx.x, tid = threadIdx.x, bs = blockDim.x;

#if __has_builtin(__builtin_amdgcn_tensor_load_to_lds) && \
    __has_builtin(__builtin_amdgcn_s_wait_tensorcnt)
  if (tid < 32) {  // one wave issues the TDM op (EXEC-independent, per-wave)
    const unsigned long long ga =
        (unsigned long long)(size_t)(seq_in + (size_t)bi * kSeq * kH);
    u32x4 g0;
    g0[0] = 1u;                                  // count=1, user descriptor
    g0[1] = (unsigned)(size_t)xs;                // lds_addr (bytes)
    g0[2] = (unsigned)(ga & 0xFFFFFFFFull);      // global_addr[31:0]
    g0[3] = (unsigned)((ga >> 32) & 0x01FFFFFFull) | (2u << 30); // addr[56:32]|type=2
    i32x8 g1;
    g1[0] = (int)((2u << 16) | (1u << 20) | (6u << 22) | (127u << 25));
    //       data_size=4B     pad_enable   pad_int=128dw pad_amt=128dw
    g1[1] = (int)((unsigned)(kH & 0xFFFF) << 16);                 // tensor_dim0 lo
    g1[2] = (int)(((unsigned)kSeq & 0xFFFFu) << 16);              // dim0 hi=0 | dim1 lo
    g1[3] = (int)((unsigned)(kH & 0xFFFF) << 16);                 // dim1 hi=0 | tile_dim0
    g1[4] = (int)(kSeq & 0xFFFF);                                 // tile_dim1 (tile_dim2=0)
    g1[5] = kH;                                                   // tensor_dim0_stride lo
    g1[6] = 0; g1[7] = 0;
    i32x4 gz = {0, 0, 0, 0};
#if __clang_major__ >= 23
    i32x8 gz8 = {0, 0, 0, 0, 0, 0, 0, 0};
    __builtin_amdgcn_tensor_load_to_lds(g0, g1, gz, gz, gz8, 0);
#else
    __builtin_amdgcn_tensor_load_to_lds(g0, g1, gz, gz, 0);
#endif
    __builtin_amdgcn_s_wait_tensorcnt(0);
  }
  __syncthreads();
#else
  for (int j = tid; j < kSeq * kH; j += bs) {
    const int s = j >> 7, c = j & (kH - 1);
    xs[s * 2 * kH + c] = seq_in[(size_t)bi * kSeq * kH + j];
  }
  __syncthreads();
#endif

  for (int l = 0; l < kL; ++l) {
    const int inw = (l == 0) ? kH : 2 * kH;
    for (int d = 0; d < 2; ++d) {
      const float* Wi = (l == 0) ? (Wih0 + (size_t)d * 4 * kH * kH)
                                 : (Wih + (size_t)((l - 1) * 2 + d) * 4 * kH * 2 * kH);
      const float* Wh = (l == 0) ? (Whh0 + (size_t)d * 4 * kH * kH)
                                 : (Whh + (size_t)((l - 1) * 2 + d) * 4 * kH * kH);
      const float* bv = (l == 0) ? (b0 + (size_t)d * 4 * kH)
                                 : (bb + (size_t)((l - 1) * 2 + d) * 4 * kH);
      for (int j = tid; j < kH; j += bs) { hcur[j] = 0.f; ccur[j] = 0.f; }
      __syncthreads();
      for (int s = 0; s < kSeq; ++s) {
        const int idx = d ? (kSeq - 1 - s) : s;
        const float* xt = xs + idx * 2 * kH;
        for (int j = tid; j < 4 * kH; j += bs) {
          float acc = bv[j];
          const float* wi = Wi + (size_t)j * inw;
          for (int k = 0; k < inw; ++k) acc += wi[k] * xt[k];
          const float* wh = Wh + (size_t)j * kH;
          for (int k = 0; k < kH; ++k) acc += wh[k] * hcur[k];
          gb[j] = acc;
        }
        __syncthreads();
        for (int j = tid; j < kH; j += bs) {
          const float cn = sigmoidf_(gb[kH + j]) * ccur[j] + sigmoidf_(gb[j]) * tanhf(gb[2 * kH + j]);
          const float hn = sigmoidf_(gb[3 * kH + j]) * tanhf(cn);
          ccur[j] = cn; hcur[j] = hn;
          (d ? hb : hf)[idx * kH + j] = hn;
        }
        __syncthreads();
      }
    }
    for (int j = tid; j < kSeq * 2 * kH; j += bs) {
      const int s = j / (2 * kH), c = j % (2 * kH);
      xs[j] = (c < kH) ? hf[s * kH + c] : hb[s * kH + c - kH];
    }
    __syncthreads();
  }
  for (int j = tid; j < 2 * kH; j += bs)
    outv[(size_t)bi * 2 * kH + j] = xs[(kSeq - 1) * 2 * kH + j];
}

// ============================================================================
// Kernel 3: ft head — softmax(relu(out@W1+b1)@W2+b2). Tiny, VALU.
// ============================================================================
__global__ void ft_head_kernel(const float* __restrict__ outv, const float* __restrict__ W1,
                               const float* __restrict__ b1, const float* __restrict__ W2,
                               const float* __restrict__ b2, float* __restrict__ dst) {
  __shared__ float hid[2 * kH];
  __shared__ float lg[kT];
  __shared__ float red[2];
  const int bi = blockIdx.x, tid = threadIdx.x;
  const float* o = outv + (size_t)bi * 2 * kH;
  if (tid < 2 * kH) {
    float acc = b1[tid];
    for (int k = 0; k < 2 * kH; ++k) acc += o[k] * W1[k * (2 * kH) + tid];
    hid[tid] = fmaxf(acc, 0.f);
  }
  __syncthreads();
  if (tid < kT) {
    float acc = b2[tid];
    for (int k = 0; k < 2 * kH; ++k) acc += hid[k] * W2[k * kT + tid];
    lg[tid] = acc;
  }
  __syncthreads();
  if (tid == 0) {
    float mx = lg[0];
    for (int j = 1; j < kT; ++j) mx = fmaxf(mx, lg[j]);
    float sm = 0.f;
    for (int j = 0; j < kT; ++j) sm += __expf(lg[j] - mx);
    red[0] = mx; red[1] = sm;
  }
  __syncthreads();
  if (tid < kT) dst[(size_t)bi * kT + tid] = __expf(lg[tid] - red[0]) / red[1];
}

// ============================================================================
// Kernel 4: mean / std heads. Tiny, VALU.
// ============================================================================
__global__ void meanstd_kernel(const float* __restrict__ outv,
                               const float* __restrict__ mW, const float* __restrict__ mb,
                               const float* __restrict__ sW, const float* __restrict__ sb,
                               float* __restrict__ mdst, float* __restrict__ sdst) {
  const int bi = blockIdx.x, tid = threadIdx.x;
  if (tid >= kP3) return;
  const float* o = outv + (size_t)bi * 2 * kH;
  float am = mb[tid], as = sb[tid];
  for (int k = 0; k < 2 * kH; ++k) { am += o[k] * mW[k * kP3 + tid]; as += o[k] * sW[k * kP3 + tid]; }
  mdst[(size_t)bi * kP3 + tid] = rintf(logf(fmaxf(am, 0.f) + 1.0f) * kParaScale - 0.5f + 1e-6f);
  sdst[(size_t)bi * kP3 + tid] = fmaxf(as, 0.01f);
}

// ============================================================================
// Kernel 5: ip heads stage 1 — hid = relu([16,256] @ W1[256,8320] + b1).
// One wave per 16x16 output tile; WMMA bf16, K=256 (8 steps).
// ============================================================================
__global__ void ip_stage1_kernel(const float* __restrict__ outv,
                                 const float* __restrict__ W1a, const float* __restrict__ b1a,
                                 const float* __restrict__ W1b, const float* __restrict__ b1b,
                                 float* __restrict__ hid) {       // [2][B][2D]
  const int h = blockIdx.y;
  const int n0 = blockIdx.x * 16;
  const int lane = threadIdx.x;
  const float* W1 = h ? W1b : W1a;
  const float* b1 = h ? b1b : b1a;
  v8f acc = {};
#pragma unroll
  for (int kk = 0; kk < (2 * kH) / 32; ++kk) {
    const int k0 = kk * 32;
    v16bf a = load_a_f32(outv, 2 * kH, 0, k0, lane, kB);
    v16bf b = load_b_f32(W1, k2D, k0, n0, lane);
    acc = wmma_bf16(a, b, acc);
  }
  const int col = n0 + (lane & 15);
  const int rbase = (lane >> 4) << 3;
  const float bias = b1[col];
#pragma unroll
  for (int i = 0; i < 8; ++i) {
    const int r = rbase + i;
    hid[((size_t)h * kB + r) * k2D + col] = fmaxf(acc[i] + bias, 0.f);
  }
}

// ============================================================================
// Kernel 6: ip heads stage 2 — logits = hid[16,8320] @ W2[8320,4160] + b2.
// Bandwidth-bound: W2 (2x138MB fp32) streamed exactly once with non-temporal
// loads; WMMA bf16, K=8320. global_prefetch_b8 hints 4 k-steps ahead.
// ============================================================================
__global__ void ip_stage2_kernel(const float* __restrict__ hid,
                                 const float* __restrict__ W2a, const float* __restrict__ b2a,
                                 const float* __restrict__ W2b, const float* __restrict__ b2b,
                                 float* __restrict__ logits) {    // [2][B][D]
  const int h = blockIdx.y;
  const int n0 = blockIdx.x * 16;
  const int lane = threadIdx.x;
  const float* W2 = h ? W2b : W2a;
  const float* b2 = h ? b2b : b2a;
  const float* A = hid + (size_t)h * kB * k2D;
  v8f acc = {};
  for (int kk = 0; kk < k2D / 32; ++kk) {
    const int k0 = kk * 32;
    if (kk + 4 < k2D / 32)
      __builtin_prefetch(W2 + (size_t)(k0 + 128 + lane) * kD + n0, 0, 1);
    v16bf a = load_a_f32(A, k2D, 0, k0, lane, kB);
    v16bf b = load_b_f32_nt(W2, kD, k0, n0, lane);
    acc = wmma_bf16(a, b, acc);
  }
  const int col = n0 + (lane & 15);
  const int rbase = (lane >> 4) << 3;
  const float bias = b2[col];
#pragma unroll
  for (int i = 0; i < 8; ++i) {
    const int r = rbase + i;
    logits[((size_t)h * kB + r) * kD + col] = acc[i] + bias;
  }
}

// ============================================================================
// Kernel 7: softmax over logits[:PVD], zero-pad to D, write into d_out.
// ============================================================================
__global__ void ip_softmax_kernel(const float* __restrict__ logits, float* __restrict__ dout) {
  __shared__ float red[256];
  const int bi = blockIdx.x, h = blockIdx.y, tid = threadIdx.x, bs = blockDim.x;
  const float* row = logits + ((size_t)h * kB + bi) * kD;
  float mx = -3.4e38f;
  for (int j = tid; j < kPVD; j += bs) mx = fmaxf(mx, row[j]);
  red[tid] = mx;
  __syncthreads();
  for (int off = bs >> 1; off > 0; off >>= 1) {
    if (tid < off) red[tid] = fmaxf(red[tid], red[tid + off]);
    __syncthreads();
  }
  mx = red[0];
  __syncthreads();
  float sm = 0.f;
  for (int j = tid; j < kPVD; j += bs) sm += __expf(row[j] - mx);
  red[tid] = sm;
  __syncthreads();
  for (int off = bs >> 1; off > 0; off >>= 1) {
    if (tid < off) red[tid] += red[tid + off];
    __syncthreads();
  }
  sm = red[0];
  float* orow = dout + (h ? OFF_I2 : OFF_I1) + (size_t)bi * kD;
  const float inv = 1.0f / sm;
  for (int j = tid; j < kD; j += bs) orow[j] = (j < kPVD) ? __expf(row[j] - mx) * inv : 0.0f;
}

// ============================================================================
extern "C" void kernel_launch(void* const* d_in, const int* in_sizes, int n_in,
                              void* d_out, int out_size, void* d_ws, size_t ws_size,
                              hipStream_t stream) {
  (void)in_sizes; (void)n_in; (void)out_size; (void)ws_size;
  // setup_inputs() order; d_in[0] (input_state) is unused by the reference
  const float* emb_s   = (const float*)d_in[1];
  const float* emb_l   = (const float*)d_in[2];
  const float* fn_emb  = (const float*)d_in[3];
  const float* fus_W1  = (const float*)d_in[4];
  const float* fus_b1  = (const float*)d_in[5];
  const float* fus_W2  = (const float*)d_in[6];
  const float* fus_b2  = (const float*)d_in[7];
  const float* cell_Wih = (const float*)d_in[8];
  const float* cell_Whh = (const float*)d_in[9];
  const float* cell_b   = (const float*)d_in[10];
  const float* bi_Wih0  = (const float*)d_in[11];
  const float* bi_Whh0  = (const float*)d_in[12];
  const float* bi_b0    = (const float*)d_in[13];
  const float* bi_Wih   = (const float*)d_in[14];
  const float* bi_Whh   = (const float*)d_in[15];
  const float* bi_b     = (const float*)d_in[16];
  const float* ft_W1    = (const float*)d_in[17];
  const float* ft_b1    = (const float*)d_in[18];
  const float* ft_W2    = (const float*)d_in[19];
  const float* ft_b2    = (const float*)d_in[20];
  const float* ip1_W1   = (const float*)d_in[21];
  const float* ip1_b1   = (const float*)d_in[22];
  const float* ip1_W2   = (const float*)d_in[23];
  const float* ip1_b2   = (const float*)d_in[24];
  const float* ip2_W1   = (const float*)d_in[25];
  const float* ip2_b1   = (const float*)d_in[26];
  const float* ip2_W2   = (const float*)d_in[27];
  const float* ip2_b2   = (const float*)d_in[28];
  const float* mean_W   = (const float*)d_in[29];
  const float* mean_b   = (const float*)d_in[30];
  const float* std_W    = (const float*)d_in[31];
  const float* std_b    = (const float*)d_in[32];
  const long long* formulas = (const long long*)d_in[33]; // numpy int64

  float* out = (float*)d_out;
  float* ws = (float*)d_ws;
  float* seq    = ws + WS_SEQ;  // [B][33][H]
  float* outv   = ws + WS_OUT;  // [B][2H]
  float* hid    = ws + WS_HID;  // [2][B][2D]
  float* logits = ws + WS_LOG;  // [2][B][D]
  bf16*  packw  = (bf16*)(ws + WS_PW); // [L][2][H][4H] bf16

  // pack cell weights into WMMA-B bf16 layout (786KB, L2-resident thereafter)
  {
    const int total = kL * 2 * kH * k4H;
    pack_cell_weights_kernel<<<(total + 255) / 256, 256, 0, stream>>>(cell_Wih, cell_Whh, packw);
  }

  // chain LSTM: state resident in LDS (~198KB dynamic; MI455X WGP has 320KB)
  const size_t smem_chain =
      (size_t)(2 * kL * kMaxN * kH + 2 * kMaxN * kH + kH + 2 * kH + 2 * kSL) * sizeof(float);
  chain_lstm_kernel<<<kB, 256, smem_chain, stream>>>(
      emb_s, emb_l, fn_emb, fus_W1, fus_b1, fus_W2, fus_b2,
      packw, cell_b, formulas, seq);

  const size_t smem_bi =
      (size_t)(kSeq * 2 * kH + 2 * kSeq * kH + 4 * kH + 2 * kH) * sizeof(float);
  bilstm_kernel<<<kB, 256, smem_bi, stream>>>(
      seq, bi_Wih0, bi_Whh0, bi_b0, bi_Wih, bi_Whh, bi_b, outv);

  ft_head_kernel<<<kB, 256, 0, stream>>>(outv, ft_W1, ft_b1, ft_W2, ft_b2, out + OFF_FT);
  meanstd_kernel<<<kB, 32, 0, stream>>>(outv, mean_W, mean_b, std_W, std_b,
                                        out + OFF_M, out + OFF_S);

  ip_stage1_kernel<<<dim3(k2D / 16, 2), 32, 0, stream>>>(outv, ip1_W1, ip1_b1, ip2_W1, ip2_b1, hid);
  ip_stage2_kernel<<<dim3(kD / 16, 2), 32, 0, stream>>>(hid, ip1_W2, ip1_b2, ip2_W2, ip2_b2, logits);
  ip_softmax_kernel<<<dim3(kB, 2), 256, 0, stream>>>(logits, out);
}
